// HeteroGNN_79740362817699
// MI455X (gfx1250) — matrix operand order repaired and checked by reading, not compiled, not run on hardware
//
#include <hip/hip_runtime.h>
#include <stdint.h>
#include <stddef.h>

// ---------------- problem constants (match reference) ----------------
#define N_ADR 200000
#define N_BAT 200000
#define N_PAR 150000
#define E_ACC 800000
#define E_DES 800000
#define E_APP 600000
#define HID   128
#define OUTF  64
#define SLOPE 0.2f

typedef __attribute__((ext_vector_type(16))) __bf16 v16bf;
typedef __attribute__((ext_vector_type(8)))  float  v8f;

// ---------------- device helpers ----------------
__device__ __forceinline__ unsigned short f2bf(float f) {
  unsigned u = __float_as_uint(f);
  u += 0x7FFFu + ((u >> 16) & 1u);           // round-to-nearest-even
  return (unsigned short)(u >> 16);
}
// monotone float<->uint mapping so atomicMax(unsigned) == float max
__device__ __forceinline__ unsigned fmap(float x) {
  unsigned u = __float_as_uint(x);
  return (u & 0x80000000u) ? ~u : (u | 0x80000000u);
}
__device__ __forceinline__ float funmap(unsigned u) {
  u = (u & 0x80000000u) ? (u ^ 0x80000000u) : ~u;
  return __uint_as_float(u);
}

// ---------------- kernels ----------------

// f32 [N,F] -> bf16 [N,Fp] with zero padding of columns F..Fp-1
__global__ void cvt_bf16_pad_kernel(const float* __restrict__ x,
                                    unsigned short* __restrict__ o,
                                    int N, int F, int Fp) {
  size_t i = (size_t)blockIdx.x * blockDim.x + threadIdx.x;
  if (i >= (size_t)N * Fp) return;
  int col = (int)(i % Fp);
  size_t row = i / Fp;
  o[i] = (col < F) ? f2bf(x[row * F + col]) : (unsigned short)0;
}

// pack f32 weight [K,N] row-major into WMMA bf16 B-operand fragments.
// layout: packed[(((kt*NT)+nt)*32 + lane)*8 + v] holds bf16 pair
//   k0 = kt*32 + (v/4)*16 + (lane/16)*8 + 2(v%4) ; n = nt*16 + lane%16  (per ISA 05_wmma)
__global__ void pack_w_kernel(const float* __restrict__ W, int Kact, int N, int NT,
                              unsigned* __restrict__ dst, int total) {
  int tid = blockIdx.x * blockDim.x + threadIdx.x;
  if (tid >= total) return;
  int v    = tid & 7;
  int lane = (tid >> 3) & 31;
  int ntkt = tid >> 8;
  int nt   = ntkt % NT;
  int kt   = ntkt / NT;
  int n  = nt * 16 + (lane & 15);
  int k0 = kt * 32 + (v >> 2) * 16 + (lane >> 4) * 8 + 2 * (v & 3);
  float lo = (k0     < Kact) ? W[(size_t)k0       * N + n] : 0.f;
  float hi = (k0 + 1 < Kact) ? W[(size_t)(k0 + 1) * N + n] : 0.f;
  dst[tid] = (unsigned)f2bf(lo) | ((unsigned)f2bf(hi) << 16);
}

// Y[Mtiles*16, Ncols] = A[.,KT*32](bf16,row-major,lda) @ Bpacked + bias (f32 out)
// Requirements: lda == KT*32 padded, M % 16 == 0.
// One wave per PAIR of 16-row M tiles: each LDS B fragment feeds 2 WMMAs.
// __launch_bounds__(256,1) raises the VGPR budget so the 2x8 v8f accumulator
// set (128 VGPRs) stays in registers (no scratch spills).
template <int NT>
__global__ void __launch_bounds__(256, 1)
gemm_wmma_kernel(const unsigned short* __restrict__ A, int lda,
                 int Mtiles, int KT,
                 const unsigned* __restrict__ Bp,
                 const float* __restrict__ bias,
                 float* __restrict__ Y, int ldy) {
  __shared__ unsigned sB[NT * 1024];           // up to KT=4 K-chunks
  const int lane = threadIdx.x & 31;
  const int wave = threadIdx.x >> 5;

  const int nB = KT * NT * 256;
  for (int i = threadIdx.x; i < nB; i += blockDim.x) sB[i] = Bp[i];
  __syncthreads();

  const int pair = blockIdx.x * (blockDim.x >> 5) + wave;
  const int mt0  = pair * 2;
  if (mt0 >= Mtiles) return;                   // wave-uniform
  const bool has1 = (mt0 + 1) < Mtiles;        // wave-uniform tail handling
  const int  mt1  = has1 ? (mt0 + 1) : mt0;    // redundant compute on tail, store guarded
  const int half = lane >> 4, l16 = lane & 15;
  const unsigned short* arow0 = A + (size_t)(mt0 * 16 + l16) * lda;
  const unsigned short* arow1 = A + (size_t)(mt1 * 16 + l16) * lda;

  union Frag { uint4 q[2]; v16bf v; };

  v8f acc0[NT], acc1[NT];
#pragma unroll
  for (int t = 0; t < NT; ++t)
#pragma unroll
    for (int j = 0; j < 8; ++j) { acc0[t][j] = 0.f; acc1[t][j] = 0.f; }

  for (int kt = 0; kt < KT; ++kt) {
    // 16-bit A 16x32 layout: VGPR v<4 -> K = half*8+2v ; v>=4 -> 16+half*8+2(v-4)
    // => two contiguous 16B runs per lane
    Frag af0, af1;
    af0.q[0] = *reinterpret_cast<const uint4*>(arow0 + kt * 32 + half * 8);
    af0.q[1] = *reinterpret_cast<const uint4*>(arow0 + kt * 32 + 16 + half * 8);
    af1.q[0] = *reinterpret_cast<const uint4*>(arow1 + kt * 32 + half * 8);
    af1.q[1] = *reinterpret_cast<const uint4*>(arow1 + kt * 32 + 16 + half * 8);

    const unsigned* bbase = sB + ((size_t)kt * NT * 256) + lane * 8;
#pragma unroll
    for (int nt = 0; nt < NT; ++nt) {
      Frag bf_;
      const unsigned* bp = bbase + nt * 256;
      bf_.q[0] = *reinterpret_cast<const uint4*>(bp);
      bf_.q[1] = *reinterpret_cast<const uint4*>(bp + 4);
      acc0[nt] = __builtin_amdgcn_wmma_f32_16x16x32_bf16(
          false, af0.v, false, bf_.v, (short)0, acc0[nt], false, false);
      acc1[nt] = __builtin_amdgcn_wmma_f32_16x16x32_bf16(
          false, af1.v, false, bf_.v, (short)0, acc1[nt], false, false);
    }
  }
  // epilogue: C layout m = r + 8*half, n = nt*16 + l16
#pragma unroll
  for (int nt = 0; nt < NT; ++nt) {
    int n = nt * 16 + l16;
    float b = bias ? bias[n] : 0.f;
#pragma unroll
    for (int r = 0; r < 8; ++r)
      Y[(size_t)(mt0 * 16 + r + 8 * half) * ldy + n] = acc0[nt][r] + b;
    if (has1) {
#pragma unroll
      for (int r = 0; r < 8; ++r)
        Y[(size_t)(mt1 * 16 + r + 8 * half) * ldy + n] = acc1[nt][r] + b;
    }
  }
}

// m = fmap(-inf), denom = 0
__global__ void init_mden_kernel(unsigned* __restrict__ m, float* __restrict__ den, int N) {
  int i = blockIdx.x * blockDim.x + threadIdx.x;
  if (i < N) { m[i] = 0x007FFFFFu; den[i] = 0.f; }
}
__global__ void zero_f32_kernel(float* __restrict__ p, size_t n) {
  size_t i = (size_t)blockIdx.x * blockDim.x + threadIdx.x;
  if (i < n) p[i] = 0.f;
}

// wave per edge: e = att . leaky_relu(hl[src] + hr[dst]); segment-max into mmax
__global__ void edge_scores_kernel(const int* __restrict__ src, const int* __restrict__ dst, int E,
                                   const float* __restrict__ hl, const float* __restrict__ hr,
                                   const float* __restrict__ att,
                                   float* __restrict__ esc, unsigned* __restrict__ mmax) {
  int lane = threadIdx.x & 31;
  int e = blockIdx.x * (blockDim.x >> 5) + (threadIdx.x >> 5);
  if (e >= E) return;
  int s = src[e], d = dst[e];
  float4 a = *reinterpret_cast<const float4*>(hl + (size_t)s * HID + lane * 4);
  float4 b = *reinterpret_cast<const float4*>(hr + (size_t)d * HID + lane * 4);
  float4 w = *reinterpret_cast<const float4*>(att + lane * 4);
  float t, acc = 0.f;
  t = a.x + b.x; t = (t > 0.f) ? t : SLOPE * t; acc += w.x * t;
  t = a.y + b.y; t = (t > 0.f) ? t : SLOPE * t; acc += w.y * t;
  t = a.z + b.z; t = (t > 0.f) ? t : SLOPE * t; acc += w.z * t;
  t = a.w + b.w; t = (t > 0.f) ? t : SLOPE * t; acc += w.w * t;
#pragma unroll
  for (int off = 16; off > 0; off >>= 1) acc += __shfl_xor(acc, off, 32);
  if (lane == 0) {
    esc[e] = acc;
    atomicMax(&mmax[d], fmap(acc));
  }
}

// thread per edge: ex = exp(e - m[dst]); denom[dst] += ex
__global__ void edge_exp_kernel(const int* __restrict__ dst, int E,
                                float* __restrict__ esc, const unsigned* __restrict__ mmax,
                                float* __restrict__ den) {
  int e = blockIdx.x * blockDim.x + threadIdx.x;
  if (e >= E) return;
  int d = dst[e];
  float ex = __expf(esc[e] - funmap(mmax[d]));
  esc[e] = ex;
  atomicAdd(&den[d], ex);
}

// wave per edge: agg[dst] += (ex/(denom+eps)) * hl[src]
__global__ void edge_scatter_kernel(const int* __restrict__ src, const int* __restrict__ dst, int E,
                                    const float* __restrict__ esc, const float* __restrict__ den,
                                    const float* __restrict__ hl, float* __restrict__ agg) {
  int lane = threadIdx.x & 31;
  int e = blockIdx.x * (blockDim.x >> 5) + (threadIdx.x >> 5);
  if (e >= E) return;
  int s = src[e], d = dst[e];
  float alpha = esc[e] / (den[d] + 1e-16f);
  float4 a = *reinterpret_cast<const float4*>(hl + (size_t)s * HID + lane * 4);
  float* o = agg + (size_t)d * HID + lane * 4;
  atomicAdd(o + 0, alpha * a.x);
  atomicAdd(o + 1, alpha * a.y);
  atomicAdd(o + 2, alpha * a.z);
  atomicAdd(o + 3, alpha * a.w);
}

// x_new(bf16) = relu(agg + bias)
__global__ void finalize_kernel(const float* __restrict__ agg, const float* __restrict__ bias,
                                size_t total, unsigned short* __restrict__ xbf) {
  size_t i = (size_t)blockIdx.x * blockDim.x + threadIdx.x;
  if (i < total) {
    float v = agg[i] + bias[i & (HID - 1)];
    xbf[i] = f2bf(fmaxf(v, 0.f));
  }
}

// ---------------- host launcher ----------------
extern "C" void kernel_launch(void* const* d_in, const int* in_sizes, int n_in,
                              void* d_out, int out_size, void* d_ws, size_t ws_size,
                              hipStream_t stream) {
  (void)in_sizes; (void)n_in; (void)out_size; (void)ws_size;

  // inputs in setup_inputs() dict order
  const float* x_in[3] = { (const float*)d_in[0], (const float*)d_in[1], (const float*)d_in[2] };
  const int* e_acc = (const int*)d_in[3];   // [2, E] : row0=src, row1=dst
  const int* e_des = (const int*)d_in[4];
  const int* e_app = (const int*)d_in[5];

  // params pytree leaves (JAX sorted-key order):
  // layers[0..1] x {acces, appartient, dessert} x {Wl, Wr, att, bias, bl, br}, then
  // lin x {adresse, batiment, parcelle} x {W, b}
  struct Gat { const float *Wl, *Wr, *att, *bias, *bl, *br; };
  Gat g[2][3];
  int pi = 6;
  for (int l = 0; l < 2; ++l)
    for (int r = 0; r < 3; ++r) {
      g[l][r].Wl   = (const float*)d_in[pi++];
      g[l][r].Wr   = (const float*)d_in[pi++];
      g[l][r].att  = (const float*)d_in[pi++];
      g[l][r].bias = (const float*)d_in[pi++];
      g[l][r].bl   = (const float*)d_in[pi++];
      g[l][r].br   = (const float*)d_in[pi++];
    }
  const float *linW[3], *linB[3];
  for (int t = 0; t < 3; ++t) { linW[t] = (const float*)d_in[pi++]; linB[t] = (const float*)d_in[pi++]; }

  // workspace carve (256B aligned)
  char* wsp = (char*)d_ws;
  auto carve = [&](size_t bytes) -> void* {
    void* p = (void*)wsp; wsp += (bytes + 255) & ~(size_t)255; return p;
  };
  const int nN[3] = { N_ADR, N_BAT, N_PAR };
  unsigned short* xbf[3];
  for (int t = 0; t < 3; ++t) xbf[t] = (unsigned short*)carve((size_t)nN[t] * HID * 2);
  float* hl  = (float*)carve((size_t)200000 * HID * 4);
  float* hr  = (float*)carve((size_t)200000 * HID * 4);
  float* agg[3];
  for (int t = 0; t < 3; ++t) agg[t] = (float*)carve((size_t)nN[t] * HID * 4);
  float*    esc  = (float*)carve((size_t)800000 * 4);
  unsigned* mmax = (unsigned*)carve((size_t)200000 * 4);
  float*    den  = (float*)carve((size_t)200000 * 4);

  auto packw = [&](const float* W, int Kact, int N) -> unsigned* {
    int KT = (Kact + 31) / 32, NT = N / 16;
    int total = KT * NT * 256;
    unsigned* dst = (unsigned*)carve((size_t)total * 4);
    pack_w_kernel<<<(total + 255) / 256, 256, 0, stream>>>(W, Kact, N, NT, dst, total);
    return dst;
  };
  // A must be K-padded to KT*32 (lda == KT*32); M must be a multiple of 16.
  auto gemm = [&](const unsigned short* A, int lda, int M, int KT,
                  const unsigned* Bp, const float* bias, float* Y, int Ncols) {
    int Mtiles = M / 16;
    int pairs  = (Mtiles + 1) / 2;
    int blocks = (pairs + 7) / 8;              // 8 waves / block
    if (Ncols == 128)
      gemm_wmma_kernel<8><<<blocks, 256, 0, stream>>>(A, lda, Mtiles, KT, Bp, bias, Y, Ncols);
    else
      gemm_wmma_kernel<4><<<blocks, 256, 0, stream>>>(A, lda, Mtiles, KT, Bp, bias, Y, Ncols);
  };

  // relation tables (pytree order: 0=acces adr->bat, 1=appartient bat->par, 2=dessert bat->adr)
  const int relSrc[3] = { 0, 1, 1 };
  const int relDst[3] = { 1, 2, 0 };
  const int relE[3]   = { E_ACC, E_APP, E_DES };
  const int* relEdge[3] = { e_acc, e_app, e_des };

  // pre-pack all weights (bf16 B fragments)
  const int F0[3]  = { 32, 64, 16 };          // true feature width per node type, layer 0
  const int Fp0[3] = { 32, 64, 32 };          // K-padded-to-32 pitch per node type, layer 0
  int Fp[3] = { Fp0[0], Fp0[1], Fp0[2] };     // current A pitch per type
  unsigned* pWl[2][3]; unsigned* pWr[2][3]; unsigned* pLin[3];
  for (int l = 0; l < 2; ++l)
    for (int r = 0; r < 3; ++r) {
      int fs = (l == 0) ? F0[relSrc[r]] : HID;
      int fd = (l == 0) ? F0[relDst[r]] : HID;
      pWl[l][r] = packw(g[l][r].Wl, fs, HID);  // pack pads K to KT*32 with zeros
      pWr[l][r] = packw(g[l][r].Wr, fd, HID);
    }
  for (int t = 0; t < 3; ++t) pLin[t] = packw(linW[t], HID, OUTF);

  // initial bf16 conversion of node features (zero-padded K)
  for (int t = 0; t < 3; ++t) {
    size_t n = (size_t)nN[t] * Fp0[t];
    cvt_bf16_pad_kernel<<<(unsigned)((n + 255) / 256), 256, 0, stream>>>(
        x_in[t], xbf[t], nN[t], F0[t], Fp0[t]);
  }

  // ----- 2 GATv2 layers -----
  for (int l = 0; l < 2; ++l) {
    for (int r = 0; r < 3; ++r) {
      int s = relSrc[r], d = relDst[r];
      int E = relE[r];
      const int* esrc = relEdge[r];
      const int* edst = relEdge[r] + E;
      // hl = x_src @ Wl + bl ; hr = x_dst @ Wr + br
      gemm(xbf[s], Fp[s], nN[s], Fp[s] / 32, pWl[l][r], g[l][r].bl, hl, HID);
      gemm(xbf[d], Fp[d], nN[d], Fp[d] / 32, pWr[l][r], g[l][r].br, hr, HID);
      // segment softmax + weighted scatter
      init_mden_kernel<<<(nN[d] + 255) / 256, 256, 0, stream>>>(mmax, den, nN[d]);
      size_t aggN = (size_t)nN[d] * HID;
      zero_f32_kernel<<<(unsigned)((aggN + 255) / 256), 256, 0, stream>>>(agg[d], aggN);
      edge_scores_kernel<<<(E + 7) / 8, 256, 0, stream>>>(esrc, edst, E, hl, hr, g[l][r].att, esc, mmax);
      edge_exp_kernel<<<(E + 255) / 256, 256, 0, stream>>>(edst, E, esc, mmax, den);
      edge_scatter_kernel<<<(E + 7) / 8, 256, 0, stream>>>(esrc, edst, E, esc, den, hl, agg[d]);
    }
    // x_t = relu(agg_t + bias) ; bias comes from the relation targeting type t
    const int biasRel[3] = { 2, 0, 1 };       // adr<-dessert, bat<-acces, par<-appartient
    for (int t = 0; t < 3; ++t) {
      size_t n = (size_t)nN[t] * HID;
      finalize_kernel<<<(unsigned)((n + 255) / 256), 256, 0, stream>>>(
          agg[t], g[l][biasRel[t]].bias, n, xbf[t]);
      Fp[t] = HID;
    }
  }

  // ----- final per-type linear heads into d_out (adresse, batiment, parcelle) -----
  float* out = (float*)d_out;
  size_t off = 0;
  for (int t = 0; t < 3; ++t) {
    gemm(xbf[t], HID, nN[t], HID / 32, pLin[t], linB[t], out + off, OUTF);
    off += (size_t)nN[t] * OUTF;
  }
}
